// MinGRU_53420803228268
// MI455X (gfx1250) — matrix operand order repaired and checked by reading, not compiled, not run on hardware
//
#include <hip/hip_runtime.h>
#include <hip/hip_bf16.h>

// ---------------- problem constants ----------------
#define B_   4
#define S_   8192
#define D_   512
#define DI_  1024
#define NC_  64            // scan chunks along S
#define CL_  (S_ / NC_)    // 128 steps per chunk
#define MROWS_ (B_ * S_)   // 32768 GEMM rows

typedef __bf16 bf16_t;
typedef __attribute__((ext_vector_type(16))) __bf16 v16bf;
typedef __attribute__((ext_vector_type(8)))  __bf16 v8bf;
typedef __attribute__((ext_vector_type(8)))  float  v8f;
typedef __attribute__((ext_vector_type(4)))  unsigned int uint4v;
typedef __attribute__((ext_vector_type(8)))  int int8v;
typedef __attribute__((ext_vector_type(4)))  int int4v;

#if defined(__HIP_DEVICE_COMPILE__) && __has_builtin(__builtin_amdgcn_tensor_load_to_lds) && __has_builtin(__builtin_amdgcn_s_wait_tensorcnt)
#define USE_TDM 1
#else
#define USE_TDM 0
#endif

// ---------------- math helpers ----------------
__device__ __forceinline__ float softplus_f(float x) {
    return fmaxf(x, 0.0f) + log1pf(__expf(-fabsf(x)));
}
__device__ __forceinline__ float logaddexp_f(float a, float b) {
    float m = fmaxf(a, b);
    if (m == -INFINITY) return -INFINITY;
    return m + log1pf(__expf(-fabsf(a - b)));
}
__device__ __forceinline__ float log_g_f(float x) {
    return (x >= 0.0f) ? __logf(x + 0.5f) : -softplus_f(-x);
}
__device__ __forceinline__ unsigned short f32_to_bf16_bits(float f) {
    union { bf16_t b; unsigned short u; } cv;
    cv.b = (bf16_t)f;
    return cv.u;
}

// ---------------- TDM: 1-D contiguous bf16 tile, global -> LDS ----------------
// D# per CDNA5 ISA. 1-D tile: tile_dim0 = nElems, tile_dim1 = 1.
// This toolchain exposes the 6-arg builtin form: (g0, g1, g2, g3, g4, cpol).
#if USE_TDM
__device__ __forceinline__ void tdm_load_1d_bf16(unsigned int ldsByteOff,
                                                 const void* gsrc,
                                                 unsigned int nElems) {
    unsigned long long ga = (unsigned long long)(size_t)gsrc;
    uint4v g0;
    g0[0] = 1u;                                            // count=1, is_restore=0, gather=0
    g0[1] = ldsByteOff;                                    // lds_addr (bytes)
    g0[2] = (unsigned int)ga;                              // global_addr[31:0]
    g0[3] = (unsigned int)((ga >> 32) & 0x01FFFFFFull)     // global_addr[56:32]
            | (2u << 30);                                  // type = 2 ("image")
    int8v g1;
    g1[0] = (int)(1u << 16);                               // workgroup_mask=0, data_size=1 (2B)
    g1[1] = (int)((nElems & 0xffffu) << 16);               // tensor_dim0[15:0]
    g1[2] = (int)((nElems >> 16) | (1u << 16));            // tensor_dim0[31:16], tensor_dim1=1
    g1[3] = (int)((nElems & 0xffffu) << 16);               // tensor_dim1 hi=0, tile_dim0=nElems
    g1[4] = (int)1;                                        // tile_dim1=1, tile_dim2=0
    g1[5] = (int)nElems;                                   // tensor_dim0_stride[31:0]
    g1[6] = 0;
    g1[7] = 0;
    int4v g2 = {0, 0, 0, 0};
    int4v g3 = {0, 0, 0, 0};
    int8v g4 = {0, 0, 0, 0, 0, 0, 0, 0};
    __builtin_amdgcn_tensor_load_to_lds(g0, g1, g2, g3, g4, 0);
}
#endif

// cooperative fallback: 256 threads, contiguous b128 copy
__device__ __forceinline__ void coop_copy_tile(unsigned short* dst,
                                               const unsigned short* src,
                                               int nElems) {
    for (int i = (int)threadIdx.x * 8; i < nElems; i += 256 * 8) {
        *(v8bf*)(dst + i) = *(const v8bf*)(src + i);
    }
}

// ---------------- WMMA fragment loaders (bf16 16x16x32) ----------------
// A (16x32 of row-major source): lanes 0-15 hold M=lane, K groups {0..7, 16..23};
// lanes 16-31 hold same M rows, K groups {8..15, 24..31}. Two 16B loads per lane.
__device__ __forceinline__ v16bf load_a_frag(const bf16_t* __restrict__ A, int lda,
                                             int mBase, int k0, int lane) {
    const int  m  = mBase + (lane & 15);
    const bool hi = lane >= 16;
    const bf16_t* row = A + (size_t)m * lda + k0;
    v8bf p0 = *(const v8bf*)(row + (hi ? 8  : 0));
    v8bf p1 = *(const v8bf*)(row + (hi ? 24 : 16));
    return __builtin_shufflevector(p0, p1, 0,1,2,3,4,5,6,7,8,9,10,11,12,13,14,15);
}

// B (32x16), logical B[k][n] = tile[n*ld + k] (tile rows are B columns, contiguous in K):
// lane holds N = lane&15; lanes 0-15: K groups {0..7, 16..23}, lanes 16-31: K groups {8..15, 24..31}.
__device__ __forceinline__ v16bf load_b_frag_lds(const unsigned short* tile, int ld,
                                                 int k0, int lane) {
    const int  n  = lane & 15;
    const bool hi = lane >= 16;
    const bf16_t* row = (const bf16_t*)tile + n * ld + k0;
    v8bf p0 = *(const v8bf*)(row + (hi ? 8  : 0));
    v8bf p1 = *(const v8bf*)(row + (hi ? 24 : 16));
    return __builtin_shufflevector(p0, p1, 0,1,2,3,4,5,6,7,8,9,10,11,12,13,14,15);
}

// ---------------- kernel 0: f32 -> bf16 conversion ----------------
__global__ __launch_bounds__(256) void cvt_bf16_kernel(const float* __restrict__ in,
                                                       unsigned short* __restrict__ out,
                                                       int n) {
    int i = blockIdx.x * 256 + threadIdx.x;
    if (i < n) out[i] = f32_to_bf16_bits(in[i]);
}

// ---------------- kernel 1: GEMM1 (X @ W_hg^T) fused with gate epilogue ----------------
// Block = 8 waves = 256-row M supertile x one 16-col N pair; each wave owns 32 rows
// (two A fragments, B fragments reused -> 4 WMMA per k-step). The (h, gate) W tiles
// (2 x 16KB) are staged once into LDS via the Tensor Data Mover and reused by all waves.
__global__ __launch_bounds__(256) void gemm1_gates_kernel(const unsigned short* __restrict__ Xb_,
                                                          const unsigned short* __restrict__ Wb_,
                                                          float* __restrict__ logC,
                                                          float* __restrict__ logV) {
    __shared__ unsigned short ldsW[2 * 16 * D_];   // 32 KB: h tile then gate tile
    const int lane  = threadIdx.x & 31;
    const int wave  = threadIdx.x >> 5;
    const int nPair = blockIdx.x & (DI_ / 16 - 1);   // 64
    const int mSup  = blockIdx.x >> 6;               // 128 supertiles of 256 rows
    const int mBase = mSup * 256 + wave * 32;
    const int nBase = nPair * 16;
    const int tileElems = 16 * D_;                   // 8192

    const unsigned short* gWh = Wb_ + (size_t)nBase * D_;          // rows [n0, n0+16)
    const unsigned short* gWg = Wb_ + (size_t)(DI_ + nBase) * D_;  // rows [Di+n0, ...)

#if USE_TDM
    if (wave == 0) {
        unsigned int ldsBase = (unsigned int)(size_t)(void*)ldsW;
        tdm_load_1d_bf16(ldsBase,                          gWh, (unsigned)tileElems);
        tdm_load_1d_bf16(ldsBase + (unsigned)tileElems*2u, gWg, (unsigned)tileElems);
        __builtin_amdgcn_s_wait_tensorcnt((unsigned short)0);
    }
#else
    coop_copy_tile(ldsW,             gWh, tileElems);
    coop_copy_tile(ldsW + tileElems, gWg, tileElems);
#endif
    __syncthreads();

    const bf16_t* Xb = (const bf16_t*)Xb_;
    v8f ch0 = {}, cg0 = {};
    v8f ch1 = {}, cg1 = {};
#pragma unroll 4
    for (int k0 = 0; k0 < D_; k0 += 32) {
        if (k0 + 32 < D_) {
            __builtin_prefetch(Xb + (size_t)(mBase +      (lane & 15)) * D_ + k0 + 32, 0, 3);
            __builtin_prefetch(Xb + (size_t)(mBase + 16 + (lane & 15)) * D_ + k0 + 32, 0, 3);
        }
        v16bf a0 = load_a_frag(Xb, D_, mBase,      k0, lane);
        v16bf a1 = load_a_frag(Xb, D_, mBase + 16, k0, lane);
        v16bf bh = load_b_frag_lds(ldsW,             D_, k0, lane);
        v16bf bg = load_b_frag_lds(ldsW + tileElems, D_, k0, lane);
        ch0 = __builtin_amdgcn_wmma_f32_16x16x32_bf16(false, a0, false, bh, (short)0, ch0, false, false);
        cg0 = __builtin_amdgcn_wmma_f32_16x16x32_bf16(false, a0, false, bg, (short)0, cg0, false, false);
        ch1 = __builtin_amdgcn_wmma_f32_16x16x32_bf16(false, a1, false, bh, (short)0, ch1, false, false);
        cg1 = __builtin_amdgcn_wmma_f32_16x16x32_bf16(false, a1, false, bg, (short)0, cg1, false, false);
    }

    const bool hiHalf = lane >= 16;
    const int  col    = nBase + (lane & 15);
#pragma unroll
    for (int half = 0; half < 2; ++half) {
        const v8f& ch = half ? ch1 : ch0;
        const v8f& cg = half ? cg1 : cg0;
        const int  mb = mBase + half * 16;
#pragma unroll
        for (int r = 0; r < 8; ++r) {
            const int   m  = mb + r + (hiHalf ? 8 : 0);
            const float h  = ch[r];
            const float g  = cg[r];
            const float lc = -softplus_f(g);                 // log_coeffs
            const float lv = -softplus_f(-g) + log_g_f(h);   // log_z + log_g(h)
            const size_t idx = (size_t)m * DI_ + col;        // [b][s][i] flat over m=b*S+s
            logC[idx] = lc;
            logV[idx] = lv;
        }
    }
}

// ---------------- kernel 2a: per-chunk scan summaries ----------------
// Composition of x -> logaddexp(a + x, v): (A,V) o (a,v) = (A+a, logaddexp(V+a, v))
__global__ __launch_bounds__(256) void scan_chunks_kernel(const float* __restrict__ logC,
                                                          const float* __restrict__ logV,
                                                          float* __restrict__ chA,
                                                          float* __restrict__ chV) {
    const int tid  = blockIdx.x * 256 + threadIdx.x;   // B*NC*DI threads
    const int i    = tid & (DI_ - 1);
    const int rest = tid >> 10;
    const int c    = rest & (NC_ - 1);
    const int b    = rest >> 6;

    float A = 0.0f, V = -INFINITY;
    size_t base = ((size_t)b * S_ + (size_t)c * CL_) * DI_ + i;
    for (int s = 0; s < CL_; ++s) {
        const float lc = logC[base];
        const float lv = logV[base];
        A += lc;
        V  = logaddexp_f(V + lc, lv);
        base += DI_;
    }
    const size_t o = ((size_t)b * NC_ + c) * DI_ + i;
    chA[o] = A;
    chV[o] = V;
}

// ---------------- kernel 2b: serial scan across chunk summaries ----------------
__global__ __launch_bounds__(256) void scan_summaries_kernel(const float* __restrict__ hidden,
                                                             const float* __restrict__ chA,
                                                             const float* __restrict__ chV,
                                                             float* __restrict__ prefix,
                                                             float* __restrict__ nextHidden) {
    const int tid = blockIdx.x * 256 + threadIdx.x;    // B*DI threads
    const int i   = tid & (DI_ - 1);
    const int b   = tid >> 10;

    float carry = __logf(hidden[(size_t)b * DI_ + i]); // log(h0), hidden in (0.1, 1.0)
    for (int c = 0; c < NC_; ++c) {
        const size_t o = ((size_t)b * NC_ + c) * DI_ + i;
        prefix[o] = carry;
        carry = logaddexp_f(carry + chA[o], chV[o]);
    }
    nextHidden[(size_t)b * DI_ + i] = __expf(carry);   // h_all[:, -1]
}

// ---------------- kernel 2c: apply prefixes, emit h_all (bf16) ----------------
__global__ __launch_bounds__(256) void apply_scan_kernel(const float* __restrict__ logC,
                                                         const float* __restrict__ logV,
                                                         const float* __restrict__ prefix,
                                                         unsigned short* __restrict__ Hb) {
    const int tid  = blockIdx.x * 256 + threadIdx.x;   // B*NC*DI threads
    const int i    = tid & (DI_ - 1);
    const int rest = tid >> 10;
    const int c    = rest & (NC_ - 1);
    const int b    = rest >> 6;

    float carry = prefix[((size_t)b * NC_ + c) * DI_ + i];
    size_t base = ((size_t)b * S_ + (size_t)c * CL_) * DI_ + i;
    for (int s = 0; s < CL_; ++s) {
        carry = logaddexp_f(carry + logC[base], logV[base]);
        Hb[base] = f32_to_bf16_bits(__expf(carry));
        base += DI_;
    }
}

// ---------------- kernel 3: GEMM2 (h_all @ W_out^T) -> out (f32) ----------------
// Block = 8 waves = 256-row M supertile x one 16-col N tile; each wave owns 32 rows.
// W_out tile (32 KB) staged via TDM into LDS, shared by all waves.
__global__ __launch_bounds__(256) void gemm2_kernel(const unsigned short* __restrict__ Hb_,
                                                    const unsigned short* __restrict__ Wob_,
                                                    float* __restrict__ out) {
    __shared__ unsigned short ldsW[16 * DI_];        // 32 KB
    const int lane  = threadIdx.x & 31;
    const int wave  = threadIdx.x >> 5;
    const int nTile = blockIdx.x & (D_ / 16 - 1);    // 32
    const int mSup  = blockIdx.x >> 5;               // 128
    const int mBase = mSup * 256 + wave * 32;
    const int nBase = nTile * 16;
    const int tileElems = 16 * DI_;                  // 16384

    const unsigned short* gW = Wob_ + (size_t)nBase * DI_;

#if USE_TDM
    if (wave == 0) {
        tdm_load_1d_bf16((unsigned int)(size_t)(void*)ldsW, gW, (unsigned)tileElems);
        __builtin_amdgcn_s_wait_tensorcnt((unsigned short)0);
    }
#else
    coop_copy_tile(ldsW, gW, tileElems);
#endif
    __syncthreads();

    const bf16_t* Hb = (const bf16_t*)Hb_;
    v8f c0 = {}, c1 = {};
#pragma unroll 4
    for (int k0 = 0; k0 < DI_; k0 += 32) {
        if (k0 + 32 < DI_) {
            __builtin_prefetch(Hb + (size_t)(mBase +      (lane & 15)) * DI_ + k0 + 32, 0, 3);
            __builtin_prefetch(Hb + (size_t)(mBase + 16 + (lane & 15)) * DI_ + k0 + 32, 0, 3);
        }
        v16bf a0 = load_a_frag(Hb, DI_, mBase,      k0, lane);
        v16bf a1 = load_a_frag(Hb, DI_, mBase + 16, k0, lane);
        v16bf b  = load_b_frag_lds(ldsW, DI_, k0, lane);
        c0 = __builtin_amdgcn_wmma_f32_16x16x32_bf16(false, a0, false, b, (short)0, c0, false, false);
        c1 = __builtin_amdgcn_wmma_f32_16x16x32_bf16(false, a1, false, b, (short)0, c1, false, false);
    }

    const bool hiHalf = lane >= 16;
    const int  col    = nBase + (lane & 15);
#pragma unroll
    for (int half = 0; half < 2; ++half) {
        const v8f& c  = half ? c1 : c0;
        const int  mb = mBase + half * 16;
#pragma unroll
        for (int r = 0; r < 8; ++r) {
            const int m = mb + r + (hiHalf ? 8 : 0);
            out[(size_t)m * D_ + col] = c[r];
        }
    }
}

// ---------------- host launcher ----------------
extern "C" void kernel_launch(void* const* d_in, const int* in_sizes, int n_in,
                              void* d_out, int out_size, void* d_ws, size_t ws_size,
                              hipStream_t stream) {
    const float* X      = (const float*)d_in[0];   // (4, 8192, 512)
    const float* hidden = (const float*)d_in[1];   // (4, 1, 1024)
    const float* W_hg   = (const float*)d_in[2];   // (2048, 512)
    const float* W_out  = (const float*)d_in[3];   // (512, 1024)
    float* out = (float*)d_out;                    // (4,8192,512) then (4,1,1024)

    const size_t nX   = (size_t)B_ * S_ * D_;      // 16,777,216
    const size_t nWhg = (size_t)(2 * DI_) * D_;    // 1,048,576
    const size_t nWo  = (size_t)D_ * DI_;          // 524,288
    const size_t nHG  = (size_t)B_ * S_ * DI_;     // 33,554,432
    const size_t nCh  = (size_t)B_ * NC_ * DI_;    // 262,144

    // workspace carve-up (256B aligned)
    char* ws = (char*)d_ws;
    size_t off = 0;
    auto carve = [&](size_t bytes) -> char* {
        char* p = ws + off;
        off = (off + bytes + 255) & ~(size_t)255;
        return p;
    };
    unsigned short* Xb     = (unsigned short*)carve(nX   * 2);
    unsigned short* Whg_b  = (unsigned short*)carve(nWhg * 2);
    unsigned short* Wout_b = (unsigned short*)carve(nWo  * 2);
    float*          logC   = (float*)carve(nHG * 4);
    float*          logV   = (float*)carve(nHG * 4);
    unsigned short* Hb     = (unsigned short*)carve(nHG * 2);
    float*          chA    = (float*)carve(nCh * 4);
    float*          chV    = (float*)carve(nCh * 4);
    float*          prefix = (float*)carve(nCh * 4);
    (void)ws_size;

    // 0) convert inputs to bf16
    cvt_bf16_kernel<<<(unsigned)(nX   / 256), 256, 0, stream>>>(X,     Xb,     (int)nX);
    cvt_bf16_kernel<<<(unsigned)(nWhg / 256), 256, 0, stream>>>(W_hg,  Whg_b,  (int)nWhg);
    cvt_bf16_kernel<<<(unsigned)(nWo  / 256), 256, 0, stream>>>(W_out, Wout_b, (int)nWo);

    // 1) GEMM1 + gate math: 128 M-supertiles x 64 N-pairs
    gemm1_gates_kernel<<<(MROWS_ / 256) * (DI_ / 16), 256, 0, stream>>>(Xb, Whg_b, logC, logV);

    // 2) chunked log-space scan
    scan_chunks_kernel   <<<(unsigned)((size_t)B_ * NC_ * DI_ / 256), 256, 0, stream>>>(logC, logV, chA, chV);
    scan_summaries_kernel<<<(unsigned)((size_t)B_ * DI_ / 256),       256, 0, stream>>>(
        hidden, chA, chV, prefix, out + (size_t)B_ * S_ * D_);
    apply_scan_kernel    <<<(unsigned)((size_t)B_ * NC_ * DI_ / 256), 256, 0, stream>>>(logC, logV, prefix, Hb);

    // 3) GEMM2: 128 M-supertiles x 32 N-tiles
    gemm2_kernel<<<(MROWS_ / 256) * (D_ / 16), 256, 0, stream>>>(Hb, Wout_b, out);
}
